// SelfAttention_24283745091744
// MI455X (gfx1250) — matrix operand order repaired and checked by reading, not compiled, not run on hardware
//
#include <hip/hip_runtime.h>

// ---------------------------------------------------------------------------
// SAGAN self-attention for MI455X (gfx1250), flash-attention style, bf16 WMMA.
//   B=8, H=W=64 -> N=4096, C=256, d=C/8=32.
//   reference:  S[i,j] = k_i . q_j ; P = softmax_j(S) ; out = gamma*(P V) + x
// All matmuls use v_wmma_f32_16x16x32_bf16 (f32 accumulate).
// ---------------------------------------------------------------------------

typedef __attribute__((ext_vector_type(16))) __bf16 v16bf;
typedef __attribute__((ext_vector_type(8)))  __bf16 v8bf;
typedef __attribute__((ext_vector_type(8)))  float  v8f;

#define BATCH 8
#define NPOS  4096
#define CH    256
#define DQK   32

// xor-shuffle within the 32-lane wave via ds_swizzle (group-of-32 mode:
// and_mask=0x1F, or=0, xor=m). Masks 1,2,4,8 stay inside each 16-lane half,
// which is exactly the lane group that shares one matrix row in C-layout.
#define SWZ_XOR_F(v, m) \
  __int_as_float(__builtin_amdgcn_ds_swizzle(__float_as_int(v), (((m) << 10) | 0x1F)))

__device__ __forceinline__ v16bf cat16(v8bf lo, v8bf hi) {
  return __builtin_shufflevector(lo, hi, 0, 1, 2, 3, 4, 5, 6, 7,
                                         8, 9, 10, 11, 12, 13, 14, 15);
}

__device__ __forceinline__ v8f wmma_bf16(v16bf a, v16bf b, v8f c) {
  return __builtin_amdgcn_wmma_f32_16x16x32_bf16(false, a, false, b,
                                                 (short)0, c, false, false);
}

// ---------------------------------------------------------------------------
// Kernel 0: pack Wq|Wk|Wv into one transposed bf16 weight matrix Wt[320][256]
// so projection B-fragments are contiguous (lane = out-channel, k contiguous).
// ---------------------------------------------------------------------------
__global__ void prep_weights_kernel(const float* __restrict__ Wq,
                                    const float* __restrict__ Wk,
                                    const float* __restrict__ Wv,
                                    __bf16* __restrict__ Wt) {
  int idx = blockIdx.x * 256 + threadIdx.x;   // 320*256 = 81920 threads
  int n = idx >> 8;                           // 0..319 output channel
  int k = idx & 255;                          // 0..255 input channel
  float v;
  if (n < 32)       v = Wq[k * 32 + n];
  else if (n < 64)  v = Wk[k * 32 + (n - 32)];
  else              v = Wv[k * 256 + (n - 64)];
  Wt[n * 256 + k] = (__bf16)v;
}

// ---------------------------------------------------------------------------
// Kernel 1: fused QKV projection GEMM.  M = B*N = 32768 rows, K = 256,
// Nout = 320 (q:0..31, k:32..63, v:64..319).  grid = (2048, 5), block = 128
// (4 waves); wave w handles n-tile blockIdx.y*4+w, one 16x16 tile, 8 WMMAs.
// Outputs: qb,kb row-major bf16 [B*N][32]; v stored TRANSPOSED vT[B][C][N].
// ---------------------------------------------------------------------------
__global__ __launch_bounds__(128) void proj_kernel(
    const float* __restrict__ x, const __bf16* __restrict__ Wt,
    const float* __restrict__ bq, const float* __restrict__ bk,
    const float* __restrict__ bv,
    __bf16* __restrict__ qb, __bf16* __restrict__ kb,
    __bf16* __restrict__ vT) {
  const int lane  = threadIdx.x & 31;
  const int w     = threadIdx.x >> 5;
  const int half  = lane >> 4;       // which 16-lane half
  const int l16   = lane & 15;
  const int ntile = blockIdx.y * 4 + w;  // 0..19
  const int mtile = blockIdx.x;          // 0..2047

  const float* xrow = x + (size_t)(mtile * 16 + l16) * CH;  // A: lane = row M
  v8f acc = {};
#pragma unroll
  for (int K0 = 0; K0 < CH; K0 += 32) {
    // A fragment (16x32 bf16): lane holds K = [8h..8h+8) and [16+8h..16+8h+8)
    float4 f0 = *(const float4*)(xrow + K0 + 8 * half);
    float4 f1 = *(const float4*)(xrow + K0 + 8 * half + 4);
    float4 f2 = *(const float4*)(xrow + K0 + 16 + 8 * half);
    float4 f3 = *(const float4*)(xrow + K0 + 16 + 8 * half + 4);
    v16bf a;
    a[0]  = (__bf16)f0.x; a[1]  = (__bf16)f0.y; a[2]  = (__bf16)f0.z; a[3]  = (__bf16)f0.w;
    a[4]  = (__bf16)f1.x; a[5]  = (__bf16)f1.y; a[6]  = (__bf16)f1.z; a[7]  = (__bf16)f1.w;
    a[8]  = (__bf16)f2.x; a[9]  = (__bf16)f2.y; a[10] = (__bf16)f2.z; a[11] = (__bf16)f2.w;
    a[12] = (__bf16)f3.x; a[13] = (__bf16)f3.y; a[14] = (__bf16)f3.z; a[15] = (__bf16)f3.w;
    // B fragment (32x16 bf16): lane = col N, K = [8h..8h+8) and [16+8h..16+8h+8)
    const __bf16* wp = Wt + (size_t)(ntile * 16 + l16) * 256 + K0;
    v16bf b = cat16(*(const v8bf*)(wp + 8 * half), *(const v8bf*)(wp + 16 + 8 * half));
    acc = wmma_bf16(a, b, acc);
  }

  const int n = ntile * 16 + l16;  // global output channel, col of this lane
  float bias = (ntile < 2) ? bq[n] : (ntile < 4 ? bk[n - 32] : bv[n - 64]);
#pragma unroll
  for (int r = 0; r < 8; ++r) {
    int rowg = mtile * 16 + r + 8 * half;     // C-layout: VGPR r -> row r/r+8
    int b_   = rowg >> 12;                    // batch (N = 4096 rows each)
    int pos  = rowg & 4095;
    __bf16 val = (__bf16)(acc[r] + bias);
    if (ntile < 2)
      qb[(size_t)(b_ * NPOS + pos) * DQK + n] = val;
    else if (ntile < 4)
      kb[(size_t)(b_ * NPOS + pos) * DQK + (n - 32)] = val;
    else
      vT[((size_t)b_ * CH + (n - 64)) * NPOS + pos] = val;   // transposed
  }
}

// ---------------------------------------------------------------------------
// Kernel 2: flash attention. Role swap per reference: attention-matrix rows
// come from k, columns from q.  grid = (32 row-blocks, 8 batches),
// block = 256 threads = 8 waves; wave owns 16 rows, 16x256 f32 accumulator.
// Per 32-column step: 2 WMMAs (S) + online softmax + 16 WMMAs (P*V).
// ---------------------------------------------------------------------------
__global__ __launch_bounds__(256) void attn_kernel(
    const float* __restrict__ x, const __bf16* __restrict__ qb,
    const __bf16* __restrict__ kb, const __bf16* __restrict__ vT,
    const float* __restrict__ gptr, float* __restrict__ out) {
  __shared__ __bf16 Vt[CH][32];        // 16 KB: v tile, [channel][position]
  __shared__ __bf16 Pb[8][16][32];     //  8 KB: per-wave P reshape buffer

  const int lane = threadIdx.x & 31;
  const int w    = threadIdx.x >> 5;
  const int half = lane >> 4;
  const int l16  = lane & 15;
  const int b    = blockIdx.y;
  const int rowBase = blockIdx.x * 128 + w * 16;

  // A fragment of k (invariant over the whole j loop): lane = row
  const __bf16* krow = kb + (size_t)(b * NPOS + rowBase + l16) * DQK;
  const v16bf afragK = cat16(*(const v8bf*)(krow + 8 * half),
                             *(const v8bf*)(krow + 16 + 8 * half));

  float m_r[8], l_r[8];
  v8f acc[16];
#pragma unroll
  for (int r = 0; r < 8; ++r) { m_r[r] = -1.0e30f; l_r[r] = 0.0f; }
#pragma unroll
  for (int t = 0; t < 16; ++t) acc[t] = v8f{};

  for (int j0 = 0; j0 < NPOS; j0 += 32) {
    // --- stage V tile (32 positions x 256 channels) transposed into LDS ---
    __syncthreads();  // previous iteration done reading Vt
    {
      const uint4* src =
          (const uint4*)(vT + ((size_t)b * CH + threadIdx.x) * NPOS + j0);
      uint4* dst = (uint4*)&Vt[threadIdx.x][0];
      dst[0] = src[0]; dst[1] = src[1]; dst[2] = src[2]; dst[3] = src[3];
    }
    __syncthreads();

    // --- S = k_rows . q_cols^T : two 16x16 tiles (K = 32) ---
    const __bf16* qrow0 = qb + (size_t)(b * NPOS + j0 + l16) * DQK;
    v16bf bq0 = cat16(*(const v8bf*)(qrow0 + 8 * half),
                      *(const v8bf*)(qrow0 + 16 + 8 * half));
    const __bf16* qrow1 = qrow0 + 16 * DQK;
    v16bf bq1 = cat16(*(const v8bf*)(qrow1 + 8 * half),
                      *(const v8bf*)(qrow1 + 16 + 8 * half));
    v8f z = {};
    v8f S0 = wmma_bf16(afragK, bq0, z);
    v8f S1 = wmma_bf16(afragK, bq1, z);

    // --- online softmax over this 32-column slab ---
    float scale[8];
#pragma unroll
    for (int r = 0; r < 8; ++r) {
      float s = fmaxf(S0[r], S1[r]);
      s = fmaxf(s, SWZ_XOR_F(s, 1));
      s = fmaxf(s, SWZ_XOR_F(s, 2));
      s = fmaxf(s, SWZ_XOR_F(s, 4));
      s = fmaxf(s, SWZ_XOR_F(s, 8));
      float mnew = fmaxf(m_r[r], s);
      float sc = __expf(m_r[r] - mnew);
      float p0 = __expf(S0[r] - mnew);
      float p1 = __expf(S1[r] - mnew);
      float rs = p0 + p1;
      rs += SWZ_XOR_F(rs, 1);
      rs += SWZ_XOR_F(rs, 2);
      rs += SWZ_XOR_F(rs, 4);
      rs += SWZ_XOR_F(rs, 8);
      l_r[r] = l_r[r] * sc + rs;
      m_r[r] = mnew;
      scale[r] = sc;
      // C-layout -> canonical [row][col] in LDS (bf16 for the next WMMA)
      int row = r + 8 * half;
      Pb[w][row][l16]      = (__bf16)p0;
      Pb[w][row][16 + l16] = (__bf16)p1;
    }
#pragma unroll
    for (int t = 0; t < 16; ++t)
#pragma unroll
      for (int r = 0; r < 8; ++r) acc[t][r] *= scale[r];

    __syncthreads();  // P visible (and Vt guaranteed resident)

    // --- A fragment of P (16x32 bf16) from LDS, two ds_load_b128s/lane ---
    v16bf pA = cat16(*(const v8bf*)&Pb[w][l16][8 * half],
                     *(const v8bf*)&Pb[w][l16][16 + 8 * half]);

    // --- O += P * V : 16 channel tiles, K = 32 ---
#pragma unroll
    for (int t = 0; t < 16; ++t) {
      const __bf16* vrow = &Vt[t * 16 + l16][8 * half];
      v16bf bv16 = cat16(*(const v8bf*)vrow, *(const v8bf*)(vrow + 16));
      acc[t] = wmma_bf16(pA, bv16, acc[t]);
    }
  }

  // --- epilogue: out = gamma * (O / l) + x ---
  const float g = *gptr;
#pragma unroll
  for (int t = 0; t < 16; ++t) {
#pragma unroll
    for (int r = 0; r < 8; ++r) {
      int pos = rowBase + r + 8 * half;
      int ch  = t * 16 + l16;
      size_t idx = ((size_t)(b * NPOS + pos)) * CH + ch;
      out[idx] = g * (acc[t][r] / l_r[r]) + x[idx];
    }
  }
}

// ---------------------------------------------------------------------------
// Launcher. Workspace layout (needs ~20.2 MB):
//   [0,2MB)   qb  bf16 [B*N][32]
//   [2,4MB)   kb  bf16 [B*N][32]
//   [4,20MB)  vT  bf16 [B][C][N]
//   [20MB,..) Wt  bf16 [320][256]
// ---------------------------------------------------------------------------
extern "C" void kernel_launch(void* const* d_in, const int* in_sizes, int n_in,
                              void* d_out, int out_size, void* d_ws,
                              size_t ws_size, hipStream_t stream) {
  const float* x  = (const float*)d_in[0];
  const float* Wq = (const float*)d_in[1];
  const float* bq = (const float*)d_in[2];
  const float* Wk = (const float*)d_in[3];
  const float* bk = (const float*)d_in[4];
  const float* Wv = (const float*)d_in[5];
  const float* bv = (const float*)d_in[6];
  const float* gm = (const float*)d_in[7];
  float* out = (float*)d_out;

  char* ws = (char*)d_ws;
  __bf16* qb = (__bf16*)(ws);
  __bf16* kb = (__bf16*)(ws + ((size_t)2 << 20));
  __bf16* vT = (__bf16*)(ws + ((size_t)4 << 20));
  __bf16* Wt = (__bf16*)(ws + ((size_t)20 << 20));

  prep_weights_kernel<<<320, 256, 0, stream>>>(Wq, Wk, Wv, Wt);
  proj_kernel<<<dim3(2048, 5), 128, 0, stream>>>(x, Wt, bq, bk, bv, qb, kb, vT);
  attn_kernel<<<dim3(32, 8), 256, 0, stream>>>(x, qb, kb, vT, gm, out);
}
